// MultiHeadAttentionEinsum_44753559224413
// MI455X (gfx1250) — matrix operand order repaired and checked by reading, not compiled, not run on hardware
//
#include <hip/hip_runtime.h>
#include <stdint.h>

typedef _Float16 half_t;
typedef __attribute__((ext_vector_type(16))) _Float16 v16h;
typedef __attribute__((ext_vector_type(8)))  float    v8f;

union Frag16 { uint32_t u[8]; v16h v; };

#define EMBED 1024
#define SEQ   2048
#define NH    16
#define HD    64
#define MTOT  4096   // B*S

// ---------------------------------------------------------------------------
// gfx1250 async global->LDS copy (ASYNCcnt) helpers
// ---------------------------------------------------------------------------
__device__ __forceinline__ uint32_t lds_addr32(const void* p) {
    // flat shared address: addr[31:0] == LDS byte offset (aperture in high bits)
    return (uint32_t)(uintptr_t)p;
}

__device__ __forceinline__ void async_b128(void* lds_dst, const void* global_src) {
    asm volatile("global_load_async_to_lds_b128 %0, %1, off"
                 :: "v"(lds_addr32(lds_dst)), "v"(global_src)
                 : "memory");
}

#if __has_builtin(__builtin_amdgcn_s_wait_asynccnt)
#define WAIT_ASYNCCNT(n) __builtin_amdgcn_s_wait_asynccnt(n)
#else
#define WAIT_ASYNCCNT(n) asm volatile("s_wait_asynccnt " #n ::: "memory")
#endif

// ---------------------------------------------------------------------------
// fp32 -> f16 convert (weights keep native [f][e] layout == B-operand [n][k])
// ---------------------------------------------------------------------------
__global__ __launch_bounds__(256) void mha_convert_w(const float* __restrict__ W,
                                                     half_t* __restrict__ Wh) {
    const size_t i = ((size_t)blockIdx.x * 256 + threadIdx.x) * 8;
    const float4 a = *(const float4*)(W + i);
    const float4 b = *(const float4*)(W + i + 4);
    union { half_t h[8]; uint4 u; } r;
    r.h[0] = (half_t)a.x; r.h[1] = (half_t)a.y; r.h[2] = (half_t)a.z; r.h[3] = (half_t)a.w;
    r.h[4] = (half_t)b.x; r.h[5] = (half_t)b.y; r.h[6] = (half_t)b.z; r.h[7] = (half_t)b.w;
    *(uint4*)(Wh + i) = r.u;
}

// ---------------------------------------------------------------------------
// GEMM: C[M=4096,N=1024] = A[fp32 MxK] * W[f16, native [n][k]]
// mode 0: f16 head-major [b,h,s,d] * scale      (Q, K projections)
// mode 2: f16 d-major    [b,h,d,s] * scale      (V projection, feeds flash V)
// mode 1: fp32 row-major [m][n] + bias          (output projection)
// Block 128x64, BK=32, 8 waves (4x2), wave tile 32x32 (2x2 WMMA frags).
// B tile staged with async global->LDS (pure linear row copies).
// ---------------------------------------------------------------------------
__global__ __launch_bounds__(256) void mha_gemm_xwt(const float*  __restrict__ A,
                                                    const half_t* __restrict__ Wh,
                                                    void*         __restrict__ Cout,
                                                    const float*  __restrict__ bias,
                                                    float scale, int mode) {
    constexpr int MT = 128, NT = 64, KT = 32;
    __shared__ __align__(16) half_t As[MT * KT];   // [m][k]
    __shared__ __align__(16) half_t Bs[NT * KT];   // [n][k]

    const int tid    = threadIdx.x;
    const int lane   = tid & 31;
    const int wid    = tid >> 5;
    const int waveM  = wid >> 1;
    const int waveN  = wid & 1;
    const int half16 = lane >> 4;
    const int l16    = lane & 15;
    const int m0     = blockIdx.y * MT;
    const int n0     = blockIdx.x * NT;

    v8f acc[2][2] = {};

    for (int k0 = 0; k0 < EMBED; k0 += KT) {
        // B tile: async copy 64 rows x 64B from Wh[n0+n][k0..k0+31]
        {
            const int nrow = tid >> 2;            // 0..63
            const int seg  = (tid & 3) * 8;       // halves (16B chunks)
            async_b128(&Bs[nrow * KT + seg],
                       Wh + (size_t)(n0 + nrow) * EMBED + k0 + seg);
        }
        // A tile: fp32 load + convert to f16 (needs registers, so not async)
        if (k0 + KT < EMBED)
            __builtin_prefetch(A + (size_t)(m0 + (tid >> 1)) * EMBED + k0 + KT, 0, 3);
#pragma unroll
        for (int i = 0; i < 4; ++i) {
            int idx = tid + i * 256;
            int row = idx >> 3;
            int cv  = (idx & 7) << 2;
            const float4 f = *(const float4*)(A + (size_t)(m0 + row) * EMBED + k0 + cv);
            half_t* dst = &As[row * KT + cv];
            dst[0] = (half_t)f.x; dst[1] = (half_t)f.y;
            dst[2] = (half_t)f.z; dst[3] = (half_t)f.w;
        }
        WAIT_ASYNCCNT(0);
        __syncthreads();

        Frag16 afr[2], bfr[2];
#pragma unroll
        for (int im = 0; im < 2; ++im) {
            const uint32_t* ap = (const uint32_t*)&As[(waveM * 32 + im * 16 + l16) * KT];
            const int pb = half16 * 4;
#pragma unroll
            for (int v = 0; v < 4; ++v) { afr[im].u[v] = ap[pb + v]; afr[im].u[4 + v] = ap[8 + pb + v]; }
        }
#pragma unroll
        for (int in = 0; in < 2; ++in) {
            const uint32_t* bp = (const uint32_t*)&Bs[(waveN * 32 + in * 16 + l16) * KT];
            const int pb = half16 * 4;
#pragma unroll
            for (int v = 0; v < 4; ++v) { bfr[in].u[v] = bp[pb + v]; bfr[in].u[4 + v] = bp[8 + pb + v]; }
        }
#pragma unroll
        for (int im = 0; im < 2; ++im)
#pragma unroll
            for (int in = 0; in < 2; ++in)
                acc[im][in] = __builtin_amdgcn_wmma_f32_16x16x32_f16(
                    false, afr[im].v, false, bfr[in].v, (short)0, acc[im][in], false, false);
        __syncthreads();
    }

#pragma unroll
    for (int im = 0; im < 2; ++im) {
#pragma unroll
        for (int in = 0; in < 2; ++in) {
            const int gn = n0 + waveN * 32 + in * 16 + l16;
#pragma unroll
            for (int r = 0; r < 8; ++r) {
                const int gm = m0 + waveM * 32 + im * 16 + r + 8 * half16;
                float val = acc[im][in][r] * scale;
                const int b = gm >> 11, s = gm & 2047;
                const int h = gn >> 6,  d = gn & 63;
                if (mode == 0) {
                    ((half_t*)Cout)[(((size_t)(b * NH + h) * SEQ + s) << 6) + d] = (half_t)val;
                } else if (mode == 2) {
                    ((half_t*)Cout)[(((size_t)(b * NH + h) * HD + d) << 11) + s] = (half_t)val;
                } else {
                    ((float*)Cout)[(size_t)gm * EMBED + gn] = val + bias[gn];
                }
            }
        }
    }
}

// ---------------------------------------------------------------------------
// Flash attention. Q head-major [bh][s][64] (pre-scaled), K head-major
// [bh][s][64], V d-major [bh][64][s]. Out fp32 [b*S+s][h*64+d].
// Grid (S/64, B*H), 128 threads (4 waves x 16 query rows).
// KV chunks of 32 double-buffered via async global->LDS copies.
// ---------------------------------------------------------------------------
__global__ __launch_bounds__(128) void mha_flash_attn(const half_t* __restrict__ Qh,
                                                      const half_t* __restrict__ Kh,
                                                      const half_t* __restrict__ Vtg,
                                                      float* __restrict__ Out) {
    __shared__ __align__(16) half_t Ks[2][32 * 64];   // [kv][d]
    __shared__ __align__(16) half_t Vt[2][64 * 32];   // [d][kv]
    __shared__ __align__(16) half_t Pls[4][16 * 32];  // per-wave P (C->A relayout)

    const int tid    = threadIdx.x;
    const int lane   = tid & 31;
    const int wid    = tid >> 5;
    const int half16 = lane >> 4;
    const int l16    = lane & 15;
    const int bh     = blockIdx.y;
    const int q0     = blockIdx.x * 64 + wid * 16;
    const size_t base = (size_t)bh * SEQ * HD;   // same for Q,K (s-major) and V (d-major)

    // Q fragments (A layout, head_dim split into two K=32 pieces)
    Frag16 qf[2];
    {
        const uint32_t* qp = (const uint32_t*)(Qh + base + (size_t)(q0 + l16) * HD);
        const int pb = half16 * 4;
#pragma unroll
        for (int v = 0; v < 4; ++v) {
            qf[0].u[v] = qp[pb + v];      qf[0].u[4 + v] = qp[8 + pb + v];
            qf[1].u[v] = qp[16 + pb + v]; qf[1].u[4 + v] = qp[24 + pb + v];
        }
    }

    float mi[8], li[8];
#pragma unroll
    for (int r = 0; r < 8; ++r) { mi[r] = -3.0e38f; li[r] = 0.0f; }
    v8f o[4] = {};

    // async issue of one 32-row KV chunk into buffer `buf` (4 ops/thread)
    auto issue_chunk = [&](int jc, int buf) {
        const int kvbase = jc * 32;
        {   // K: one contiguous 4KB block
            const char* g = (const char*)(Kh + base + (size_t)kvbase * HD);
            char* l = (char*)&Ks[buf][0];
            const int off = tid * 32;
            async_b128(l + off,      g + off);
            async_b128(l + off + 16, g + off + 16);
        }
        {   // V: 64 rows of 64B, row stride S halves in global, linear in LDS
            const int d   = tid >> 1;
            const int seg = (tid & 1) * 32;   // bytes
            const char* g = (const char*)(Vtg + base + (size_t)d * SEQ + kvbase);
            char* l = (char*)&Vt[buf][d * 32];
            async_b128(l + seg,      g + seg);
            async_b128(l + seg + 16, g + seg + 16);
        }
    };

    issue_chunk(0, 0);

    for (int jc = 0; jc < SEQ / 32; ++jc) {
        const int cur = jc & 1;
        if (jc + 1 < SEQ / 32) {
            issue_chunk(jc + 1, cur ^ 1);
            WAIT_ASYNCCNT(4);   // current chunk's 4 ops complete (in-order)
        } else {
            WAIT_ASYNCCNT(0);
        }
        __syncthreads();

        // Scores S[16x32]: two 16x16 C-frags, K-dim = head_dim (2 WMMA each)
        v8f s0 = {}, s1 = {};
#pragma unroll
        for (int dh = 0; dh < 2; ++dh) {
            Frag16 b0, b1;
            const uint32_t* kp0 = (const uint32_t*)&Ks[cur][(l16) * 64];
            const uint32_t* kp1 = (const uint32_t*)&Ks[cur][(16 + l16) * 64];
            const int pb = dh * 16 + half16 * 4;
#pragma unroll
            for (int v = 0; v < 4; ++v) { b0.u[v] = kp0[pb + v]; b1.u[v] = kp1[pb + v]; b0.u[4 + v] = kp0[8 + pb + v]; b1.u[4 + v] = kp1[8 + pb + v]; }
            s0 = __builtin_amdgcn_wmma_f32_16x16x32_f16(false, qf[dh].v, false, b0.v, (short)0, s0, false, false);
            s1 = __builtin_amdgcn_wmma_f32_16x16x32_f16(false, qf[dh].v, false, b1.v, (short)0, s1, false, false);
        }

        // Online softmax (row = r + 8*half16, cols spread over 16-lane half)
#pragma unroll
        for (int r = 0; r < 8; ++r) {
            float mx = fmaxf(s0[r], s1[r]);
#pragma unroll
            for (int off = 8; off >= 1; off >>= 1) mx = fmaxf(mx, __shfl_xor(mx, off, 32));
            float mnew = fmaxf(mi[r], mx);
            float fct  = __expf(mi[r] - mnew);
            float p0   = __expf(s0[r] - mnew);
            float p1   = __expf(s1[r] - mnew);
            float rs   = p0 + p1;
#pragma unroll
            for (int off = 8; off >= 1; off >>= 1) rs += __shfl_xor(rs, off, 32);
            li[r] = li[r] * fct + rs;
            mi[r] = mnew;
            o[0][r] *= fct; o[1][r] *= fct; o[2][r] *= fct; o[3][r] *= fct;
            const int mrow = r + 8 * half16;
            Pls[wid][mrow * 32 + l16]      = (half_t)p0;
            Pls[wid][mrow * 32 + 16 + l16] = (half_t)p1;
        }

        // P (C layout) -> A-frag via per-wave LDS bounce, then O += P @ V
        Frag16 pf;
        {
            const uint32_t* pp = (const uint32_t*)&Pls[wid][l16 * 32];
            const int pb = half16 * 4;
#pragma unroll
            for (int v = 0; v < 4; ++v) { pf.u[v] = pp[pb + v]; pf.u[4 + v] = pp[8 + pb + v]; }
        }
#pragma unroll
        for (int t = 0; t < 4; ++t) {
            Frag16 vf;
            const uint32_t* vp = (const uint32_t*)&Vt[cur][(t * 16 + l16) * 32];
            const int pb = half16 * 4;
#pragma unroll
            for (int v = 0; v < 4; ++v) { vf.u[v] = vp[pb + v]; vf.u[4 + v] = vp[8 + pb + v]; }
            o[t] = __builtin_amdgcn_wmma_f32_16x16x32_f16(false, pf.v, false, vf.v, (short)0, o[t], false, false);
        }
        __syncthreads();   // all waves done with buffers before next async overwrite
    }

    const int b = bh >> 4, h = bh & 15;
#pragma unroll
    for (int t = 0; t < 4; ++t) {
#pragma unroll
        for (int r = 0; r < 8; ++r) {
            int s = q0 + r + 8 * half16;
            int d = t * 16 + l16;
            Out[(size_t)(b * SEQ + s) * EMBED + h * HD + d] = o[t][r] / li[r];
        }
    }
}

// ---------------------------------------------------------------------------
extern "C" void kernel_launch(void* const* d_in, const int* in_sizes, int n_in,
                              void* d_out, int out_size, void* d_ws, size_t ws_size,
                              hipStream_t stream) {
    const float* xq = (const float*)d_in[0];
    const float* xk = (const float*)d_in[1];
    const float* xv = (const float*)d_in[2];
    const float* Wq = (const float*)d_in[3];
    const float* Wk = (const float*)d_in[4];
    const float* Wv = (const float*)d_in[5];
    const float* Wo = (const float*)d_in[6];
    const float* bo = (const float*)d_in[7];

    char* ws = (char*)d_ws;
    const size_t WT_BYTES   = (size_t)EMBED * EMBED * sizeof(half_t);   // 2 MB
    const size_t PROJ_BYTES = (size_t)MTOT * EMBED * sizeof(half_t);    // 8 MB
    half_t* WhQ = (half_t*)(ws + 0 * WT_BYTES);
    half_t* WhK = (half_t*)(ws + 1 * WT_BYTES);
    half_t* WhV = (half_t*)(ws + 2 * WT_BYTES);
    half_t* WhO = (half_t*)(ws + 3 * WT_BYTES);
    half_t* Qh  = (half_t*)(ws + 4 * WT_BYTES);
    half_t* Kh  = (half_t*)(ws + 4 * WT_BYTES + 1 * PROJ_BYTES);
    half_t* Vtg = (half_t*)(ws + 4 * WT_BYTES + 2 * PROJ_BYTES);
    float*  Ao  = (float*) (ws + 4 * WT_BYTES + 3 * PROJ_BYTES);        // fp32, 16 MB

    const int cblocks = (EMBED * EMBED) / (256 * 8);
    mha_convert_w<<<cblocks, 256, 0, stream>>>(Wq, WhQ);
    mha_convert_w<<<cblocks, 256, 0, stream>>>(Wk, WhK);
    mha_convert_w<<<cblocks, 256, 0, stream>>>(Wv, WhV);
    mha_convert_w<<<cblocks, 256, 0, stream>>>(Wo, WhO);

    dim3 ggrid(EMBED / 64, MTOT / 128);
    const float qscale = 0.125f;   // 1/sqrt(HEAD_DIM)
    mha_gemm_xwt<<<ggrid, 256, 0, stream>>>(xq, WhQ, (void*)Qh,  nullptr, qscale, 0);
    mha_gemm_xwt<<<ggrid, 256, 0, stream>>>(xk, WhK, (void*)Kh,  nullptr, 1.0f,   0);
    mha_gemm_xwt<<<ggrid, 256, 0, stream>>>(xv, WhV, (void*)Vtg, nullptr, 1.0f,   2);

    dim3 fgrid(SEQ / 64, 2 * NH);
    mha_flash_attn<<<fgrid, 128, 0, stream>>>(Qh, Kh, Vtg, Ao);

    mha_gemm_xwt<<<ggrid, 256, 0, stream>>>(Ao, WhO, d_out, bo, 1.0f, 1);
}
